// CacaAttention_8478265442829
// MI455X (gfx1250) — matrix operand order repaired and checked by reading, not compiled, not run on hardware
//
#include <hip/hip_runtime.h>
#include <cstdint>
#include <cstddef>

#define S_LEN 2048
#define HID   2048
#define NH    16
#define NKV   4
#define HD    128
#define WIN   1024
#define QN    (NH * HD)   // 2048
#define KN    (NKV * HD)  // 512

typedef __attribute__((ext_vector_type(16))) __bf16 bf16x16;
typedef __attribute__((ext_vector_type(8)))  float  f32x8;

union FragB { bf16x16 v; unsigned u[8]; };

// 16-bit A-matrix 16x32 K-index map (ISA 7.12.2): lanes 0-15 -> K 0..7,16..23; lanes 16-31 -> +8
__device__ __forceinline__ int kpairA(int j, int lane) {
  return ((j >> 2) << 4) | (((lane >> 4) & 1) << 3) | ((j & 3) << 1);
}
// B-matrix 32x16 K-index map (same K order as A: lanes 0-15 -> K 0..7,16..23; lanes 16-31 -> +8)
__device__ __forceinline__ int kpairB(int j, int lane) {
  return ((j >> 2) << 4) | (((lane >> 4) & 1) << 3) | ((j & 3) << 1);
}

__device__ __forceinline__ f32x8 wmma_bf16(const FragB& a, const FragB& b, f32x8 c) {
  return __builtin_amdgcn_wmma_f32_16x16x32_bf16(false, a.v, false, b.v,
                                                 (short)0, c, false, false);
}

// CDNA5 async global->LDS copy (per-lane 16B), tracked on ASYNCcnt.
// lds = LDS byte offset (low 32 bits of generic shared address), ga = 64-bit global address.
__device__ __forceinline__ void async_ld_b128(unsigned lds, uint64_t ga) {
  asm volatile("global_load_async_to_lds_b128 %0, %1, off"
               :: "v"(lds), "v"(ga) : "memory");
}
__device__ __forceinline__ void wait_async0() {
  asm volatile("s_wait_asynccnt 0x0" ::: "memory");
}
__device__ __forceinline__ unsigned lds_off(const void* p) {
  return (unsigned)(uintptr_t)p;   // LDS generic addr low 32 bits == LDS offset
}

// ---------------- elementwise converts ----------------
__global__ void cvt_f32_to_bf16(const float* __restrict__ x, __bf16* __restrict__ y, int n) {
  int i = blockIdx.x * blockDim.x + threadIdx.x;
  if (i < n) y[i] = (__bf16)x[i];
}

// RoPE: out[i] = x[i]*cos(pos*invf) + rot_half(x)[i]*sin(pos*invf), head-local d in [0,128)
__global__ void rope_f32_to_bf16(const float* __restrict__ x, __bf16* __restrict__ y,
                                 int rowW, int n) {
  int i = blockIdx.x * blockDim.x + threadIdx.x;
  if (i >= n) return;
  int row = i / rowW;
  int rem = i - row * rowW;
  int d = rem & (HD - 1);
  int j = d & 63;
  // inv_freq = 10000^(-j/64) = exp(-j * ln(10000)/64)
  float inv = expf(-(float)j * (9.210340371976184f / 64.0f));
  float ang = (float)row * inv;
  float sn, cs;
  sincosf(ang, &sn, &cs);
  float xv = x[i];
  float xr = (d < 64) ? -x[i + 64] : x[i - 64];
  y[i] = (__bf16)(xv * cs + xr * sn);
}

// ---------------- tiled bf16 GEMM: C(f32) = A(bf16, MxK) * B(bf16, KxN) ----------------
// block 128 threads = 4 waves; block tile 64x64; BK=32; wave tile 32x32 = 2x2 WMMA
__global__ __launch_bounds__(128) void gemm_bf16_f32(
    const __bf16* __restrict__ A, const __bf16* __restrict__ B,
    float* __restrict__ C, int M, int N, int K) {
  __shared__ __bf16 At[64][40];   // +8 pad
  __shared__ __bf16 Bt[32][72];   // +8 pad
  const int tid  = threadIdx.x;
  const int lane = tid & 31;
  const int wv   = tid >> 5;
  const int wm   = wv >> 1, wn = wv & 1;
  const int bm   = blockIdx.y * 64, bn = blockIdx.x * 64;
  const int n15  = lane & 15;
  f32x8 acc[2][2] = {};
  for (int k0 = 0; k0 < K; k0 += 32) {
    // async global->LDS staging: 2x A(16B) + 2x B(16B) per thread
#pragma unroll
    for (int l = 0; l < 2; ++l) {
      int idx = (tid + l * 128) * 8;
      int ar = idx >> 5, acol = idx & 31;
      async_ld_b128(lds_off(&At[ar][acol]),
                    (uint64_t)(uintptr_t)&A[(size_t)(bm + ar) * K + k0 + acol]);
      int br = idx >> 6, bcol = idx & 63;
      async_ld_b128(lds_off(&Bt[br][bcol]),
                    (uint64_t)(uintptr_t)&B[(size_t)(k0 + br) * N + bn + bcol]);
      // speculative prefetch of next K tile of A (dropped if OOB)
      __builtin_prefetch(&A[(size_t)(bm + ar) * K + k0 + 32 + acol], 0, 1);
    }
    wait_async0();
    __syncthreads();
    FragB af[2], bfr[2];
#pragma unroll
    for (int im = 0; im < 2; ++im)
#pragma unroll
      for (int j = 0; j < 8; ++j)
        af[im].u[j] = *(const unsigned*)&At[wm * 32 + im * 16 + n15][kpairA(j, lane)];
#pragma unroll
    for (int in_ = 0; in_ < 2; ++in_)
#pragma unroll
      for (int j = 0; j < 8; ++j) {
        int kp = kpairB(j, lane);
        int nn = wn * 32 + in_ * 16 + n15;
        unsigned lo = *(const unsigned short*)&Bt[kp][nn];
        unsigned hi = *(const unsigned short*)&Bt[kp + 1][nn];
        bfr[in_].u[j] = lo | (hi << 16);
      }
#pragma unroll
    for (int im = 0; im < 2; ++im)
#pragma unroll
      for (int in_ = 0; in_ < 2; ++in_)
        acc[im][in_] = wmma_bf16(af[im], bfr[in_], acc[im][in_]);
    __syncthreads();
  }
  const int half8 = ((lane >> 4) & 1) << 3;
#pragma unroll
  for (int im = 0; im < 2; ++im)
#pragma unroll
    for (int in_ = 0; in_ < 2; ++in_) {
      float* cp = C + (size_t)(bm + wm * 32 + im * 16 + half8) * N
                    + (bn + wn * 32 + in_ * 16 + n15);
#pragma unroll
      for (int r = 0; r < 8; ++r)
        cp[(size_t)r * N] = acc[im][in_][r];
    }
}

// ---------------- sliding-window flash attention, one wave per (head, 16-query tile) ----
__global__ __launch_bounds__(32) void attn_swa(
    const __bf16* __restrict__ Qb, const __bf16* __restrict__ Kb,
    const __bf16* __restrict__ Vb, __bf16* __restrict__ Ob) {
  const int qt   = blockIdx.x;       // 0..127
  const int h    = blockIdx.y;       // 0..15
  const int hk   = h >> 2;           // GQA: 4 query heads per kv head
  const int lane = threadIdx.x;
  const int q0   = qt * 16;
  const int n15  = lane & 15;
  const int half8 = ((lane >> 4) & 1) << 3;

  __shared__ __bf16 Qt[16][136];
  __shared__ __bf16 Kt[32][136];
  __shared__ __bf16 Vt[32][136];
  __shared__ __bf16 Pt[16][40];

  // stage Q tile (16 x 128) via async copy, then build 4 A-fragments held for the loop
  {
    int row = lane >> 1;
    int c0  = (lane & 1) * 64;
    const __bf16* src = Qb + (size_t)(q0 + row) * QN + h * HD + c0;
    unsigned ldsQ = lds_off(&Qt[row][c0]);
#pragma unroll
    for (int c = 0; c < 64; c += 8)
      async_ld_b128(ldsQ + 2u * c, (uint64_t)(uintptr_t)(src + c));
    wait_async0();
  }
  FragB qf[4];
#pragma unroll
  for (int s4 = 0; s4 < 4; ++s4)
#pragma unroll
    for (int j = 0; j < 8; ++j)
      qf[s4].u[j] = *(const unsigned*)&Qt[n15][s4 * 32 + kpairA(j, lane)];

  f32x8 acc[8] = {};
  float mrow[8], lrow[8];
#pragma unroll
  for (int r = 0; r < 8; ++r) { mrow[r] = -1e30f; lrow[r] = 0.f; }

  int kStart = q0 - WIN; if (kStart < 0) kStart = 0; kStart &= ~31;
  const int kEnd = q0 + 15;   // inclusive; keys > q masked by dist<0

  for (int kc = kStart; kc <= kEnd; kc += 32) {
    // async K/V tile staging: lane owns row kc+lane (clamped rows get masked)
    {
      int krow = kc + lane; if (krow > S_LEN - 1) krow = S_LEN - 1;
      const __bf16* ks = Kb + (size_t)krow * KN + hk * HD;
      const __bf16* vs = Vb + (size_t)krow * KN + hk * HD;
      unsigned ldsK = lds_off(&Kt[lane][0]);
      unsigned ldsV = lds_off(&Vt[lane][0]);
#pragma unroll
      for (int c = 0; c < 128; c += 8) {
        async_ld_b128(ldsK + 2u * c, (uint64_t)(uintptr_t)(ks + c));
        async_ld_b128(ldsV + 2u * c, (uint64_t)(uintptr_t)(vs + c));
      }
      wait_async0();
    }

    // scores: two 16x16 tiles (32 keys), each = sum of 4 WMMAs over d
    float sc[2][8];
#pragma unroll
    for (int t = 0; t < 2; ++t) {
      f32x8 sv = {};
#pragma unroll
      for (int s4 = 0; s4 < 4; ++s4) {
        FragB kf;
#pragma unroll
        for (int j = 0; j < 8; ++j)
          kf.u[j] = *(const unsigned*)&Kt[t * 16 + n15][s4 * 32 + kpairB(j, lane)];
        sv = wmma_bf16(qf[s4], kf, sv);
      }
#pragma unroll
      for (int r = 0; r < 8; ++r) {
        int q   = q0 + r + half8;
        int key = kc + t * 16 + n15;
        int dist = q - key;
        float v = sv[r] * 0.08838834764831845f;   // 1/sqrt(128)
        sc[t][r] = (dist < 0 || dist > WIN) ? -1e30f : v;
      }
    }

    // online softmax (rows live across 16-lane halves -> width-16 reductions)
#pragma unroll
    for (int r = 0; r < 8; ++r) {
      float rm = fmaxf(sc[0][r], sc[1][r]);
#pragma unroll
      for (int off = 1; off < 16; off <<= 1)
        rm = fmaxf(rm, __shfl_xor(rm, off, 16));
      float mnew  = fmaxf(mrow[r], rm);
      float alpha = __expf(mrow[r] - mnew);   // exp(0)=1 while row still fully masked
      mrow[r] = mnew;
      float p0 = __expf(sc[0][r] - mnew);
      float p1 = __expf(sc[1][r] - mnew);
      float rs = p0 + p1;
#pragma unroll
      for (int off = 1; off < 16; off <<= 1)
        rs += __shfl_xor(rs, off, 16);
      lrow[r] = lrow[r] * alpha + rs;
#pragma unroll
      for (int f = 0; f < 8; ++f) acc[f][r] *= alpha;
      int m = r + half8;
      Pt[m][n15]      = (__bf16)p0;
      Pt[m][16 + n15] = (__bf16)p1;
    }

    // P (C-layout) -> A-fragment via LDS; same-wave DS ops are in-order
    FragB pf;
#pragma unroll
    for (int j = 0; j < 8; ++j)
      pf.u[j] = *(const unsigned*)&Pt[n15][kpairA(j, lane)];

    // PV: 8 WMMAs (d in 8 tiles of 16)
#pragma unroll
    for (int f = 0; f < 8; ++f) {
      FragB vf;
#pragma unroll
      for (int j = 0; j < 8; ++j) {
        int kp = kpairB(j, lane);
        unsigned lo = *(const unsigned short*)&Vt[kp][f * 16 + n15];
        unsigned hi = *(const unsigned short*)&Vt[kp + 1][f * 16 + n15];
        vf.u[j] = lo | (hi << 16);
      }
      acc[f] = wmma_bf16(pf, vf, acc[f]);
    }
  }

  // normalize and write attention output (bf16) in [S][NH*HD]
#pragma unroll
  for (int r = 0; r < 8; ++r) {
    float inv = 1.0f / lrow[r];
    int row = q0 + r + half8;
#pragma unroll
    for (int f = 0; f < 8; ++f) {
      int col = h * HD + f * 16 + n15;
      Ob[(size_t)row * QN + col] = (__bf16)(acc[f][r] * inv);
    }
  }
}

// ---------------- launch ----------------
extern "C" void kernel_launch(void* const* d_in, const int* in_sizes, int n_in,
                              void* d_out, int out_size, void* d_ws, size_t ws_size,
                              hipStream_t stream) {
  const float* hs = (const float*)d_in[0];
  const float* wq = (const float*)d_in[1];
  const float* wk = (const float*)d_in[2];
  const float* wv = (const float*)d_in[3];
  const float* wo = (const float*)d_in[4];
  float* out = (float*)d_out;
  char* ws = (char*)d_ws;

  size_t off = 0;
  auto take = [&](size_t bytes) -> char* {
    char* p = ws + off;
    off += (bytes + 255) & ~(size_t)255;
    return p;
  };
  __bf16* Xb  = (__bf16*)take((size_t)S_LEN * HID * 2);
  __bf16* Wqb = (__bf16*)take((size_t)HID * QN * 2);
  __bf16* Wkb = (__bf16*)take((size_t)HID * KN * 2);
  __bf16* Wvb = (__bf16*)take((size_t)HID * KN * 2);
  __bf16* Wob = (__bf16*)take((size_t)QN * HID * 2);
  float*  Qf  = (float*)take((size_t)S_LEN * QN * 4);
  float*  Kf  = (float*)take((size_t)S_LEN * KN * 4);
  float*  Vf  = (float*)take((size_t)S_LEN * KN * 4);
  __bf16* Qb  = (__bf16*)take((size_t)S_LEN * QN * 2);
  __bf16* Kb  = (__bf16*)take((size_t)S_LEN * KN * 2);
  __bf16* Vb  = (__bf16*)take((size_t)S_LEN * KN * 2);
  __bf16* Ob  = (__bf16*)take((size_t)S_LEN * QN * 2);

  auto cvt = [&](const float* x, __bf16* y, int n) {
    cvt_f32_to_bf16<<<(n + 255) / 256, 256, 0, stream>>>(x, y, n);
  };
  cvt(hs, Xb, S_LEN * HID);
  cvt(wq, Wqb, HID * QN);
  cvt(wk, Wkb, HID * KN);
  cvt(wv, Wvb, HID * KN);
  cvt(wo, Wob, QN * HID);

  dim3 blk(128);
  gemm_bf16_f32<<<dim3(QN / 64, S_LEN / 64), blk, 0, stream>>>(Xb, Wqb, Qf, S_LEN, QN, HID);
  gemm_bf16_f32<<<dim3(KN / 64, S_LEN / 64), blk, 0, stream>>>(Xb, Wkb, Kf, S_LEN, KN, HID);
  gemm_bf16_f32<<<dim3(KN / 64, S_LEN / 64), blk, 0, stream>>>(Xb, Wvb, Vf, S_LEN, KN, HID);

  rope_f32_to_bf16<<<(S_LEN * QN + 255) / 256, 256, 0, stream>>>(Qf, Qb, QN, S_LEN * QN);
  rope_f32_to_bf16<<<(S_LEN * KN + 255) / 256, 256, 0, stream>>>(Kf, Kb, KN, S_LEN * KN);
  cvt(Vf, Vb, S_LEN * KN);

  attn_swa<<<dim3(S_LEN / 16, NH), 32, 0, stream>>>(Qb, Kb, Vb, Ob);

  gemm_bf16_f32<<<dim3(HID / 64, S_LEN / 64), blk, 0, stream>>>(Ob, Wob, out, S_LEN, HID, QN);
}
